// ConsciousnessClassifier_88218628260364
// MI455X (gfx1250) — compile-verified, matrix operand order checked by reading, not run on hardware
//
#include <hip/hip_runtime.h>
#include <hip/hip_bf16.h>

// ---------------------------------------------------------------------------
// ConsciousnessClassifier on MI455X (gfx1250, wave32, WMMA).
// conv2 (31.7 of ~34 GFLOP) and FC1 run on v_wmma_f32_16x16x32_f16 with f32
// accumulation.  conv2 uses a tap-pair K-ordering so B-fragments are two
// contiguous 16B LDS loads (ds_load_b128); FC1 fragments are float4 loads.
// ---------------------------------------------------------------------------

typedef __attribute__((ext_vector_type(16))) _Float16 v16h;
typedef __attribute__((ext_vector_type(8)))  _Float16 v8h;
typedef __attribute__((ext_vector_type(8)))  float    v8f;
typedef __attribute__((ext_vector_type(4)))  float    v4f;

#define C_CH   173
#define L_IN   400
#define F1     16
#define F2     32
#define K1     9
#define K2     7
#define KIM    (F1 * K2)             // 112
#define POOL   10
#define WIN    40
#define ACT_ROW (C_CH * F2 * POOL)   // 55360
#define B_SZ   64
#define EPSV   1e-5f
#define Y1ROWS (L_IN + 8)            // 3 zero rows front, 5 back (tap<=7)

// 16-bit A/B matrix 16x32 lane/VGPR layout (ISA 7.12.2):
//   element e (0..15), lane-half hi = lane>=16:
//   K = 2*(j&3) + 8*hi + 16*(j>=4) + (e&1),  j = e>>1
// => per lane the 16 elements are two 8-consecutive K runs:
//    e=0..7  -> K = 8*hi   .. 8*hi+7
//    e=8..15 -> K = 16+8*hi.. 16+8*hi+7
__device__ __forceinline__ int klocal16(int e, int hi) {
    int j = e >> 1, elem = e & 1;
    return ((j & 3) << 1) + (hi << 3) + ((j >> 2) << 4) + elem;
}

// ===========================================================================
// Kernel 1: per (channel, batch) tower.
// conv1 scalar -> Y1 f16 position-major in LDS ->
// conv2 as 4 tap-pair WMMA k-steps -> BN/ReLU -> maxpool(40)
// ===========================================================================
__global__ __launch_bounds__(128) void cc_tower_kernel(
    const float* __restrict__ x,
    const float* __restrict__ w1, const float* __restrict__ b1,
    const float* __restrict__ g1, const float* __restrict__ beta1,
    const float* __restrict__ m1, const float* __restrict__ v1,
    const float* __restrict__ w2, const float* __restrict__ b2,
    const float* __restrict__ g2, const float* __restrict__ beta2,
    const float* __restrict__ m2, const float* __restrict__ v2,
    float* __restrict__ act)
{
    __shared__ float      xs[L_IN];
    __shared__ float      w1s[F1 * K1];
    __shared__ float      sc1[F1], sh1[F1];
    __shared__ float      sc2[F2], sh2[F2];
    __shared__ _Float16   w2h[F2 * KIM];          // conv2 weights, f16
    __shared__ _Float16   y1t[Y1ROWS * F1];       // [pos+3][ci], zero-padded
    __shared__ unsigned   pool[F2 * POOL];        // maxpool accum (vals >= 0)

    const int c   = blockIdx.x;      // channel
    const int b   = blockIdx.y;      // batch
    const int tid = threadIdx.x;
    const int lane = tid & 31;
    const int wave = tid >> 5;
    const int hi   = lane >> 4;
    const int lm   = lane & 15;

    // ---- stage inputs / params ------------------------------------------
    for (int i = tid; i < L_IN; i += 128)
        xs[i] = x[((size_t)b * C_CH + c) * L_IN + i];
    for (int i = tid; i < F1 * K1; i += 128)
        w1s[i] = w1[(size_t)c * F1 * K1 + i];
    for (int i = tid; i < F2 * KIM; i += 128)
        w2h[i] = (_Float16)w2[(size_t)c * F2 * KIM + i];
    if (tid < F1) {
        int cf = c * F1 + tid;
        float inv = g1[cf] * rsqrtf(v1[cf] + EPSV);
        sc1[tid] = inv;
        sh1[tid] = (b1[cf] - m1[cf]) * inv + beta1[cf];
    }
    if (tid < F2) {
        int cf = c * F2 + tid;
        float inv = g2[cf] * rsqrtf(v2[cf] + EPSV);
        sc2[tid] = inv;
        sh2[tid] = (b2[cf] - m2[cf]) * inv + beta2[cf];
    }
    // zero halo rows of y1t: rows 0..2 and 403..407
    if (tid < 3 * F1)              y1t[tid] = (_Float16)0.f;
    if (tid < 5 * F1)              y1t[(L_IN + 3) * F1 + tid] = (_Float16)0.f;
    for (int i = tid; i < F2 * POOL; i += 128)
        pool[i] = 0u;              // post-ReLU values are >= 0
    __syncthreads();

    // ---- conv1 (1->16, k=9, pad=4) + BN + ReLU, scalar ------------------
    // p-major / f-inner so the transposed stores are contiguous halves.
    for (int idx = tid; idx < F1 * L_IN; idx += 128) {
        int f = idx & 15, p = idx >> 4;
        float s = 0.f;
        #pragma unroll
        for (int t = 0; t < K1; ++t) {
            int q = p + t - 4;
            if (q >= 0 && q < L_IN) s += xs[q] * w1s[f * K1 + t];
        }
        float v = fmaxf(s * sc1[f] + sh1[f], 0.f);
        y1t[(p + 3) * F1 + f] = (_Float16)v;
    }
    __syncthreads();

    // ---- conv2 as WMMA, tap-pair K-ordering -----------------------------
    // k-step ks covers taps (2ks, 2ks+1):  K<16 -> ci=K,    tap=2ks
    //                                      K>=16 -> ci=K-16, tap=2ks+1
    // A fragments hoisted: 2 m-tiles x 4 k-steps (tap 7 zero-padded).
    v16h afr[2][4];
    #pragma unroll
    for (int mt = 0; mt < 2; ++mt)
        #pragma unroll
        for (int ks = 0; ks < 4; ++ks) {
            #pragma unroll
            for (int e = 0; e < 16; ++e) {
                int kl  = klocal16(e, hi);
                int ci  = kl & 15;
                int tap = 2 * ks + (kl >> 4);
                _Float16 v = (_Float16)0.f;
                if (tap < K2) v = w2h[(mt * 16 + lm) * KIM + ci * K2 + tap];
                afr[mt][ks][e] = v;
            }
        }

    for (int nt = wave; nt < L_IN / 16; nt += 4) {   // 25 position tiles
        const int nrow = nt * 16 + lm;               // output position n
        v8f c0 = {}; v8f c1 = {};
        #pragma unroll
        for (int ks = 0; ks < 4; ++ks) {
            // B-frag: two contiguous 16B LDS loads (rows n+2ks, n+2ks+1;
            // +3 pad folded into the row index: row = n + tap)
            v8h blo = *(const v8h*)&y1t[(nrow + 2 * ks    ) * F1 + 8 * hi];
            v8h bhi = *(const v8h*)&y1t[(nrow + 2 * ks + 1) * F1 + 8 * hi];
            v16h bf = __builtin_shufflevector(blo, bhi,
                        0,1,2,3,4,5,6,7,8,9,10,11,12,13,14,15);
            c0 = __builtin_amdgcn_wmma_f32_16x16x32_f16(
                     false, afr[0][ks], false, bf, (short)0, c0, false, false);
            c1 = __builtin_amdgcn_wmma_f32_16x16x32_f16(
                     false, afr[1][ks], false, bf, (short)0, c1, false, false);
        }
        // epilogue: BN2 + ReLU + maxpool(40) via LDS atomic max on bits
        int widx = nrow / WIN;
        #pragma unroll
        for (int r = 0; r < 8; ++r) {
            int m0 = hi * 8 + r;                 // f32 C/D layout: M=8*hi+r
            float v0 = fmaxf(c0[r] * sc2[m0] + sh2[m0], 0.f);
            atomicMax(&pool[m0 * POOL + widx], __float_as_uint(v0));
            int m1i = 16 + m0;
            float v1x = fmaxf(c1[r] * sc2[m1i] + sh2[m1i], 0.f);
            atomicMax(&pool[m1i * POOL + widx], __float_as_uint(v1x));
        }
    }
    __syncthreads();

    // ---- write pooled features: act[b][c*320 + f*10 + w] ----------------
    for (int i = tid; i < F2 * POOL; i += 128)
        act[(size_t)b * ACT_ROW + (size_t)c * (F2 * POOL) + i] =
            __uint_as_float(pool[i]);
}

// ===========================================================================
// Kernel 2a: h[64][128] = bc1 broadcast (atomic-add target)
// ===========================================================================
__global__ __launch_bounds__(256) void cc_init_h_kernel(
    float* __restrict__ h, const float* __restrict__ bc1)
{
    int i = blockIdx.x * 256 + threadIdx.x;
    if (i < B_SZ * 128) h[i] = bc1[i & 127];
}

// ===========================================================================
// Kernel 2b: FC1 partials via WMMA.  grid = (8 n-tiles, 10 k-chunks),
// 4 waves = 4 m-tiles.  5536 K per chunk = 173 k-steps of 32.
// Per lane each fragment = two 8-float runs -> four float4 loads.
// ===========================================================================
__global__ __launch_bounds__(128) void cc_fc1_kernel(
    const float* __restrict__ act, const float* __restrict__ wc1,
    float* __restrict__ h)
{
    const int nt   = blockIdx.x;          // 0..7
    const int kc   = blockIdx.y;          // 0..9
    const int lane = threadIdx.x & 31;
    const int wave = threadIdx.x >> 5;    // m-tile 0..3
    const int hi   = lane >> 4;
    const int lm   = lane & 15;
    const int m    = wave * 16 + lm;
    const int n    = nt * 16 + lm;

    const v4f* a4 = (const v4f*)(act + (size_t)m * ACT_ROW);
    const v4f* w4 = (const v4f*)(wc1 + (size_t)n * ACT_ROW);
    const int  base = kc * (5536 / 4) + 2 * hi;   // v4f units: (kbase+8*hi)/4

    v8f acc = {};
    for (int ks = 0; ks < 173; ++ks) {
        int i0 = base + ks * 8;                   // (k0 + 8*hi)/4
        v4f A0 = a4[i0], A1 = a4[i0 + 1], A2 = a4[i0 + 4], A3 = a4[i0 + 5];
        v4f B0 = w4[i0], B1 = w4[i0 + 1], B2 = w4[i0 + 4], B3 = w4[i0 + 5];
        v16h a, bf;
        #pragma unroll
        for (int e = 0; e < 4; ++e) {
            a[e]       = (_Float16)A0[e];
            a[4 + e]   = (_Float16)A1[e];
            a[8 + e]   = (_Float16)A2[e];
            a[12 + e]  = (_Float16)A3[e];
            bf[e]      = (_Float16)B0[e];
            bf[4 + e]  = (_Float16)B1[e];
            bf[8 + e]  = (_Float16)B2[e];
            bf[12 + e] = (_Float16)B3[e];
        }
        acc = __builtin_amdgcn_wmma_f32_16x16x32_f16(
                  false, a, false, bf, (short)0, acc, false, false);
    }
    #pragma unroll
    for (int r = 0; r < 8; ++r) {
        int mo = wave * 16 + hi * 8 + r;
        atomicAdd(&h[mo * 128 + n], acc[r]);      // global_atomic_add_f32
    }
}

// ===========================================================================
// Kernel 3: out[b] = bc2 + sum_n relu(h[b][n]) * wc2[n]
// ===========================================================================
__global__ __launch_bounds__(64) void cc_head_kernel(
    const float* __restrict__ h, const float* __restrict__ wc2,
    const float* __restrict__ bc2, float* __restrict__ out)
{
    int bi = threadIdx.x;
    if (bi < B_SZ) {
        float s = bc2[0];
        #pragma unroll 4
        for (int n = 0; n < 128; ++n)
            s += fmaxf(h[bi * 128 + n], 0.f) * wc2[n];
        out[bi] = s;
    }
}

// ===========================================================================
extern "C" void kernel_launch(void* const* d_in, const int* in_sizes, int n_in,
                              void* d_out, int out_size, void* d_ws, size_t ws_size,
                              hipStream_t stream)
{
    (void)in_sizes; (void)n_in; (void)out_size; (void)ws_size;
    const float* x     = (const float*)d_in[0];
    const float* w1    = (const float*)d_in[1];
    const float* b1    = (const float*)d_in[2];
    const float* g1    = (const float*)d_in[3];
    const float* beta1 = (const float*)d_in[4];
    const float* m1    = (const float*)d_in[5];
    const float* v1    = (const float*)d_in[6];
    const float* w2    = (const float*)d_in[7];
    const float* b2    = (const float*)d_in[8];
    const float* g2    = (const float*)d_in[9];
    const float* beta2 = (const float*)d_in[10];
    const float* m2    = (const float*)d_in[11];
    const float* v2    = (const float*)d_in[12];
    const float* wc1   = (const float*)d_in[13];
    const float* bc1   = (const float*)d_in[14];
    const float* wc2   = (const float*)d_in[15];
    const float* bc2   = (const float*)d_in[16];

    float* act = (float*)d_ws;                                  // 64*55360*4 B
    float* h   = (float*)((char*)d_ws +
                          (size_t)B_SZ * ACT_ROW * sizeof(float)); // 64*128*4 B

    cc_tower_kernel<<<dim3(C_CH, B_SZ), 128, 0, stream>>>(
        x, w1, b1, g1, beta1, m1, v1, w2, b2, g2, beta2, m2, v2, act);

    cc_init_h_kernel<<<dim3((B_SZ * 128 + 255) / 256), 256, 0, stream>>>(h, bc1);

    cc_fc1_kernel<<<dim3(8, 10), 128, 0, stream>>>(act, wc1, h);

    cc_head_kernel<<<dim3(1), 64, 0, stream>>>(h, wc2, bc2, (float*)d_out);
}
